// DynamicGNN_68083821576968
// MI455X (gfx1250) — matrix operand order repaired and checked by reading, not compiled, not run on hardware
//
#include <hip/hip_runtime.h>
#include <hip/hip_bf16.h>
#include <stdint.h>

#define BATCH   32
#define NNODES  1024
#define DIM     256
#define HID     1024
#define LAYERS  4

typedef __attribute__((ext_vector_type(16))) __bf16        v16bf;
typedef __attribute__((ext_vector_type(8)))  float         v8f;
typedef __attribute__((ext_vector_type(4)))  unsigned int  v4u;

// ---------- helpers ----------

__device__ __forceinline__ unsigned short f2bf(float f) {
  unsigned u = __builtin_bit_cast(unsigned, f);
  unsigned rnd = 0x7FFFu + ((u >> 16) & 1u);
  return (unsigned short)((u + rnd) >> 16);
}

// A-fragment: 16x32 bf16 from row-major source: two 16B runs per lane -> 2x b128.
__device__ __forceinline__ v16bf load_a_frag(const unsigned short* base, int ld) {
  const int lane = threadIdx.x & 31;
  const unsigned int* p = (const unsigned int*)(base + (size_t)(lane & 15) * ld);
  const int o = (lane >> 4) << 2;
  v4u lo = *(const v4u*)(p + o);
  v4u hi = *(const v4u*)(p + 8 + o);
  return __builtin_bit_cast(v16bf, __builtin_shufflevector(lo, hi, 0, 1, 2, 3, 4, 5, 6, 7));
}

// B-fragment from TRANSPOSED source bT[n][k]: two 16B runs per lane -> 2x b128.
__device__ __forceinline__ v16bf load_bT_frag(const unsigned short* baseT, int ldt) {
  const int lane = threadIdx.x & 31;
  const unsigned int* p =
      (const unsigned int*)(baseT + (size_t)(lane & 15) * ldt) + ((lane >> 4) << 2);
  v4u lo = *(const v4u*)(p);
  v4u hi = *(const v4u*)(p + 8);
  return __builtin_bit_cast(v16bf, __builtin_shufflevector(lo, hi, 0, 1, 2, 3, 4, 5, 6, 7));
}

__device__ __forceinline__ v8f wmma_bf16(v16bf a, v16bf b, v8f c) {
  return __builtin_amdgcn_wmma_f32_16x16x32_bf16(false, a, false, b, (short)0, c,
                                                 false, false);
}

// ---------- kernels ----------

// fp32 [L][K][N] -> bf16 transposed [L][N][K]
__global__ void k_cvtT(const float* __restrict__ in, unsigned short* __restrict__ outT,
                       int K, int N) {
  int i = blockIdx.x * 256 + threadIdx.x;
  int n = i % N;
  int t = i / N;
  int k = t % K;
  int l = t / K;
  outT[((size_t)l * N + n) * K + k] = f2bf(in[i]);
}

// h = pos @ embed_w + embed_b ; fp32 master + bf16 mirror (both [b][n][d])
__global__ void k_embed(const float* __restrict__ pos, const float* __restrict__ ew,
                        const float* __restrict__ eb, float* __restrict__ h32,
                        unsigned short* __restrict__ hbf) {
  int idx = blockIdx.x * blockDim.x + threadIdx.x;
  int d = idx & (DIM - 1);
  int bn = idx >> 8;
  float px = pos[bn * 2 + 0], py = pos[bn * 2 + 1];
  float v = px * ew[d] + py * ew[DIM + d] + eb[d];
  h32[idx] = v;
  hbf[idx] = f2bf(v);
}

// Transpose hbf [b][n][d] -> hbfT [b][d][n]. Tile staged with the CDNA5
// async global->LDS datapath (ASYNCcnt) instead of VGPR round trips.
__global__ __launch_bounds__(256) void k_transT(const unsigned short* __restrict__ in,
                                                unsigned short* __restrict__ out) {
  __shared__ __align__(8) unsigned short tile[32][34];  // pad 34: b32-aligned, bank-spread
  const int n0 = blockIdx.x * 32, d0 = blockIdx.y * 32, b = blockIdx.z;
  const int tid = threadIdx.x;
#pragma unroll
  for (int e = tid; e < 32 * 16; e += 256) {
    const int r = e >> 4, c = e & 15;  // 4 bytes (2 elems) per async op
    const unsigned short* gp = in + ((size_t)b * NNODES + n0 + r) * DIM + d0 + 2 * c;
    unsigned lds = (unsigned)((r * 34 + 2 * c) * 2);  // 'tile' is at LDS offset 0
    unsigned long long ga = (unsigned long long)gp;
    asm volatile("global_load_async_to_lds_b32 %0, %1, off"
                 :
                 : "v"(lds), "v"(ga)
                 : "memory");
  }
  asm volatile("s_wait_asynccnt 0x0" ::: "memory");
  __syncthreads();
  for (int e = tid; e < 32 * 32; e += 256) {
    const int dd = e >> 5, nn = e & 31;
    out[((size_t)b * DIM + d0 + dd) * NNODES + n0 + nn] = tile[nn][dd];
  }
}

// A[b,i,:] = softmax_j(-0.5*||p_i-p_j||^2) in bf16 (row max is exactly 0).
__global__ __launch_bounds__(256) void k_adj(const float* __restrict__ pos,
                                             unsigned short* __restrict__ Abf) {
  const int b = blockIdx.x >> 7;
  const int rowblk = blockIdx.x & 127;
  __shared__ float sx[NNODES], sy[NNODES];
  for (int j = threadIdx.x; j < NNODES; j += 256) {
    sx[j] = pos[((size_t)b * NNODES + j) * 2 + 0];
    sy[j] = pos[((size_t)b * NNODES + j) * 2 + 1];
  }
  __syncthreads();
  const int wave = threadIdx.x >> 5, lane = threadIdx.x & 31;
  const int i = rowblk * 8 + wave;
  const float xi = sx[i], yi = sy[i];
  float ev[32];
  float sum = 0.f;
#pragma unroll
  for (int t = 0; t < 32; t++) {
    int j = t * 32 + lane;
    float dx = xi - sx[j], dy = yi - sy[j];
    float e = __expf(-0.5f * (dx * dx + dy * dy));
    ev[t] = e;
    sum += e;
  }
#pragma unroll
  for (int off = 16; off; off >>= 1) sum += __shfl_xor(sum, off, 32);
  const float inv = 1.0f / sum;
  unsigned short* row = Abf + ((size_t)b * NNODES + i) * NNODES;
#pragma unroll
  for (int t = 0; t < 32; t++) row[t * 32 + lane] = f2bf(ev[t] * inv);
}

// m[b] = A[b] @ h[b]. 64x256 tile/WG; 8 waves = 4 row-groups x 2 col-groups.
// Depth-4 B rotation + K unrolled by 2 with A register ping-pong (no movs).
__global__ __launch_bounds__(256) void k_gemm_Ah(const unsigned short* __restrict__ Abf,
                                                 const unsigned short* __restrict__ hbfT,
                                                 unsigned short* __restrict__ mbf) {
  const int b = blockIdx.x >> 4;
  const int r0 = (blockIdx.x & 15) * 64;
  const int wave = threadIdx.x >> 5, lane = threadIdx.x & 31;
  const int rg = wave & 3, cg = wave >> 2;
  const int hf = lane >> 4, nl = lane & 15;

  v8f acc[8];
  v8f z = {};
#pragma unroll
  for (int i = 0; i < 8; i++) acc[i] = z;

  const unsigned short* arow = Abf + ((size_t)b * NNODES + r0 + 16 * rg) * NNODES;
  const unsigned short* bT =
      hbfT + (size_t)b * DIM * NNODES + (size_t)(128 * cg) * NNODES;

  v16bf bfb[4];
#pragma unroll
  for (int t = 0; t < 4; t++)
    bfb[t] = load_bT_frag(bT + (size_t)(t * 16) * NNODES, NNODES);

  auto step = [&](v16bf af, int kk) {
#pragma unroll
    for (int nb = 0; nb < 8; nb++) {
      acc[nb] = wmma_bf16(af, bfb[nb & 3], acc[nb]);
      // refill slot with fragment 4 ahead: (kk, nb+4) or (kk+1, nb-4)
      const int nbn = (nb + 4) & 7;
      const int kkn = (nb >= 4) ? ((kk + 1) & 31) : kk;
      bfb[nb & 3] = load_bT_frag(bT + (size_t)(nbn * 16) * NNODES + kkn * 32, NNODES);
    }
  };

  v16bf afA = load_a_frag(arow, NNODES);
  v16bf afB = load_a_frag(arow + 32, NNODES);
  for (int kk = 0; kk < 32; kk += 2) {
    step(afA, kk);
    afA = load_a_frag(arow + ((kk + 2) & 31) * 32, NNODES);  // wraps at tail
    step(afB, kk + 1);
    afB = load_a_frag(arow + ((kk + 3) & 31) * 32, NNODES);
  }

  unsigned short* mrow = mbf + ((size_t)b * NNODES + r0 + 16 * rg) * DIM;
#pragma unroll
  for (int nb = 0; nb < 8; nb++) {
#pragma unroll
    for (int v = 0; v < 8; v++) {
      int m = v + 8 * hf;
      mrow[(size_t)m * DIM + 128 * cg + nb * 16 + nl] = f2bf(acc[nb][v]);
    }
  }
}

// Fused: h32 += relu(m@w1+b1)@w2 + b2 ; hbf refreshed. 64x1024 bf16 ReLU tile
// lives in LDS (132 KB of the 320 KB WGP LDS). Depth-4 B rotation both phases;
// phase-2 K loop unrolled by 2 with A register ping-pong.
__global__ __launch_bounds__(256) void k_mlp(const unsigned short* __restrict__ mbf,
                                             const unsigned short* __restrict__ w1T,
                                             const float* __restrict__ b1,
                                             const unsigned short* __restrict__ w2T,
                                             const float* __restrict__ b2,
                                             float* __restrict__ h32,
                                             unsigned short* __restrict__ hbf) {
  const int b = blockIdx.x >> 4;
  const int r0 = (blockIdx.x & 15) * 64;
  const int wave = threadIdx.x >> 5, lane = threadIdx.x & 31;
  const int rg = wave & 3, cg = wave >> 2;
  const int hf = lane >> 4, nl = lane & 15;

  __shared__ __align__(16) unsigned short act[64 * 1032];  // ld 1032 (2064B = 129*16)

  // ---- phase 1: act = relu(m_tile @ w1 + b1) ----
  const unsigned short* mrow = mbf + ((size_t)b * NNODES + r0 + 16 * rg) * DIM;
  v16bf afr[8];  // K=256 of A resident, reused for all 32 col blocks
#pragma unroll
  for (int kk = 0; kk < 8; kk++) afr[kk] = load_a_frag(mrow + kk * 32, DIM);

  const unsigned short* w1c = w1T + (size_t)(512 * cg) * DIM;
  v16bf bfb[4];
#pragma unroll
  for (int t = 0; t < 4; t++) bfb[t] = load_bT_frag(w1c + t * 32, DIM);

  for (int nb = 0; nb < 32; nb++) {
    const int n0 = 512 * cg + nb * 16;
    v8f a0 = {}, a1 = {};  // split accumulator chain
#pragma unroll
    for (int kk = 0; kk < 8; kk++) {
      if (kk & 1) a1 = wmma_bf16(afr[kk], bfb[kk & 3], a1);
      else        a0 = wmma_bf16(afr[kk], bfb[kk & 3], a0);
      // refill with fragment 4 ahead: (nb, kk+4) or (nb+1, kk-4)
      const int kkn = (kk + 4) & 7;
      const int nbn = (nb + (kk >= 4 ? 1 : 0)) & 31;
      bfb[kk & 3] = load_bT_frag(w1c + (size_t)(nbn * 16) * DIM + kkn * 32, DIM);
    }
    const float bias = b1[n0 + nl];
#pragma unroll
    for (int v = 0; v < 8; v++) {
      int m = 16 * rg + v + 8 * hf;
      act[(size_t)m * 1032 + n0 + nl] = f2bf(fmaxf(a0[v] + a1[v] + bias, 0.f));
    }
  }
  __syncthreads();

  // ---- phase 2: h += act @ w2 + b2 ----
  v8f acc2[8];
  v8f z = {};
#pragma unroll
  for (int i = 0; i < 8; i++) acc2[i] = z;
  const unsigned short* actrow = act + (size_t)(16 * rg) * 1032;
  const unsigned short* w2c = w2T + (size_t)(128 * cg) * HID;

  v16bf bfb2[4];
#pragma unroll
  for (int t = 0; t < 4; t++)
    bfb2[t] = load_bT_frag(w2c + (size_t)(t * 16) * HID, HID);

  auto step2 = [&](v16bf af, int kk) {
#pragma unroll
    for (int nb = 0; nb < 8; nb++) {
      acc2[nb] = wmma_bf16(af, bfb2[nb & 3], acc2[nb]);
      const int nbn = (nb + 4) & 7;
      const int kkn = (nb >= 4) ? ((kk + 1) & 31) : kk;
      bfb2[nb & 3] = load_bT_frag(w2c + (size_t)(nbn * 16) * HID + kkn * 32, HID);
    }
  };

  v16bf afA = load_a_frag(actrow, 1032);
  v16bf afB = load_a_frag(actrow + 32, 1032);
  for (int kk = 0; kk < 32; kk += 2) {
    step2(afA, kk);
    afA = load_a_frag(actrow + ((kk + 2) & 31) * 32, 1032);
    step2(afB, kk + 1);
    afB = load_a_frag(actrow + ((kk + 3) & 31) * 32, 1032);
  }

#pragma unroll
  for (int nb = 0; nb < 8; nb++) {
    const int n0 = 128 * cg + nb * 16;
    const float bias = b2[n0 + nl];
#pragma unroll
    for (int v = 0; v < 8; v++) {
      int row = r0 + 16 * rg + v + 8 * hf;
      size_t idx = ((size_t)b * NNODES + row) * DIM + n0 + nl;
      float nv = h32[idx] + acc2[nb][v] + bias;
      h32[idx] = nv;
      hbf[idx] = f2bf(nv);
    }
  }
}

// two-stage deterministic pooling
__global__ void k_pool1(const float* __restrict__ h32, float* __restrict__ part) {
  int b = blockIdx.x >> 3, seg = blockIdx.x & 7, d = threadIdx.x;
  float s = 0.f;
  for (int n = seg * 128; n < (seg + 1) * 128; n++)
    s += h32[((size_t)b * NNODES + n) * DIM + d];
  part[((size_t)seg * BATCH + b) * DIM + d] = s;
}
__global__ void k_pool2(const float* __restrict__ part, float* __restrict__ pooled) {
  int b = blockIdx.x, d = threadIdx.x;
  float s = 0.f;
#pragma unroll
  for (int seg = 0; seg < 8; seg++) s += part[((size_t)seg * BATCH + b) * DIM + d];
  pooled[b * DIM + d] = s * (1.0f / (float)NNODES);
}

// t = relu(pooled @ r1_w + r1_b)  [32,1024], fp32
__global__ void k_read1(const float* __restrict__ pooled, const float* __restrict__ r1w,
                        const float* __restrict__ r1b, float* __restrict__ t) {
  int i = blockIdx.x * blockDim.x + threadIdx.x;
  int b = i >> 10, n = i & 1023;
  float s = r1b[n];
  for (int k = 0; k < DIM; k++) s += pooled[b * DIM + k] * r1w[k * HID + n];
  t[i] = fmaxf(s, 0.f);
}

// out = t @ r2_w + r2_b  [32,1024], fp32
__global__ void k_read2(const float* __restrict__ t, const float* __restrict__ r2w,
                        const float* __restrict__ r2b, float* __restrict__ out) {
  int i = blockIdx.x * blockDim.x + threadIdx.x;
  int b = i >> 10, n = i & 1023;
  float s = r2b[n];
  for (int k = 0; k < HID; k++) s += t[b * HID + k] * r2w[k * HID + n];
  out[i] = s;
}

// ---------- host ----------

extern "C" void kernel_launch(void* const* d_in, const int* in_sizes, int n_in,
                              void* d_out, int out_size, void* d_ws, size_t ws_size,
                              hipStream_t stream) {
  const float* pos  = (const float*)d_in[0];
  const float* ew   = (const float*)d_in[1];
  const float* eb   = (const float*)d_in[2];
  const float* w1   = (const float*)d_in[3];
  const float* b1   = (const float*)d_in[4];
  const float* w2   = (const float*)d_in[5];
  const float* b2   = (const float*)d_in[6];
  const float* r1w  = (const float*)d_in[7];
  const float* r1b  = (const float*)d_in[8];
  const float* r2w  = (const float*)d_in[9];
  const float* r2b  = (const float*)d_in[10];
  float* out = (float*)d_out;

  char* ws = (char*)d_ws;
  size_t off = 0;
  float*          h32  = (float*)(ws + off);          off += (size_t)BATCH * NNODES * DIM * 4;
  unsigned short* hbf  = (unsigned short*)(ws + off); off += (size_t)BATCH * NNODES * DIM * 2;
  unsigned short* hbfT = (unsigned short*)(ws + off); off += (size_t)BATCH * NNODES * DIM * 2;
  unsigned short* Abf  = (unsigned short*)(ws + off); off += (size_t)BATCH * NNODES * NNODES * 2;
  unsigned short* mbf  = (unsigned short*)(ws + off); off += (size_t)BATCH * NNODES * DIM * 2;
  unsigned short* w1T  = (unsigned short*)(ws + off); off += (size_t)LAYERS * DIM * HID * 2;
  unsigned short* w2T  = (unsigned short*)(ws + off); off += (size_t)LAYERS * HID * DIM * 2;
  float*          part = (float*)(ws + off);          off += (size_t)8 * BATCH * DIM * 4;
  float*          pool = (float*)(ws + off);          off += (size_t)BATCH * DIM * 4;
  float*          tbuf = (float*)(ws + off);          off += (size_t)BATCH * HID * 4;

  const int nW = LAYERS * DIM * HID;
  k_cvtT<<<(nW + 255) / 256, 256, 0, stream>>>(w1, w1T, DIM, HID);  // w1T: [L][H][D]
  k_cvtT<<<(nW + 255) / 256, 256, 0, stream>>>(w2, w2T, HID, DIM);  // w2T: [L][D][H]

  k_embed<<<(BATCH * NNODES * DIM) / 256, 256, 0, stream>>>(pos, ew, eb, h32, hbf);
  k_adj<<<BATCH * (NNODES / 8), 256, 0, stream>>>(pos, Abf);

  dim3 tgrid(NNODES / 32, DIM / 32, BATCH), tblk(256, 1);
  for (int l = 0; l < LAYERS; l++) {
    k_transT<<<tgrid, tblk, 0, stream>>>(hbf, hbfT);
    k_gemm_Ah<<<BATCH * (NNODES / 64), 256, 0, stream>>>(Abf, hbfT, mbf);
    k_mlp<<<BATCH * (NNODES / 64), 256, 0, stream>>>(
        mbf, w1T + (size_t)l * DIM * HID, b1 + (size_t)l * HID,
        w2T + (size_t)l * HID * DIM, b2 + (size_t)l * DIM, h32, hbf);
  }

  k_pool1<<<BATCH * 8, DIM, 0, stream>>>(h32, part);
  k_pool2<<<BATCH, DIM, 0, stream>>>(part, pool);
  k_read1<<<(BATCH * HID) / 256, 256, 0, stream>>>(pool, r1w, r1b, tbuf);
  k_read2<<<(BATCH * HID) / 256, 256, 0, stream>>>(tbuf, r2w, r2b, out);
}
